// mustang_pooling_v1_36163624632873
// MI455X (gfx1250) — compile-verified, matrix operand order checked by reading, not run on hardware
//
#include <hip/hip_runtime.h>

typedef __attribute__((ext_vector_type(16))) __bf16 bf16x16;
typedef __attribute__((ext_vector_type(8)))  float  f32x8;

#define NN     50000
#define EE     800000
#define K0N    35000
#define K1N    24500
#define DIN0   1072
#define DIN1   112
#define PAD0   1152      // multiple of KC (128)
#define PAD1   128
#define SORTP  65536
#define KC     128       // bf16 elements of K staged in LDS per chunk

// ---------------- workspace layout (bytes) ----------------
// HB region (115.2MB) is reused after GEMM consumes it.
#define OFF_HB     ((size_t)0)            // bf16 H : up to 50000*1152*2 = 115.2MB
#define OFF_ACC    ((size_t)12000000)     // f32 acc : n*128*4  (inside HB region, after GEMM)
#define OFF_HF     ((size_t)40000000)     // f32 hfeat : n*64*4
#define OFF_AGG    ((size_t)56000000)     // f32 agg : n*64*4
#define OFF_WT     ((size_t)116000000)    // bf16 W^T : 256*Kpad*2
#define OFF_XLR    ((size_t)117000000)    // f32 XLR : n*256*4 (xl cols 0..127, xr 128..255)
#define OFF_XN1    ((size_t)169000000)    // f32 pooled x layer0 : 35000*64*4
#define OFF_XN2    ((size_t)178000000)    // f32 pooled x layer1 : 24500*64*4
#define OFF_RWPE1  ((size_t)185000000)    // f32 : 35000*16*4
#define OFF_NAT1   ((size_t)188000000)    // int : 35000
#define OFF_SRC    ((size_t)189000000)    // int : E
#define OFF_DST    ((size_t)193000000)    // int : E
#define OFF_ALV    ((size_t)197000000)    // int : E
#define OFF_MX     ((size_t)201000000)    // u32 : n*2 (ordered-encoded float)
#define OFF_DEN    ((size_t)202000000)    // f32 : n*2
#define OFF_SCORE  ((size_t)203000000)    // f32 : n
#define OFF_KEYS   ((size_t)204000000)    // u64 : 65536
#define OFF_PERM   ((size_t)205000000)    // int : knew
#define OFF_SVAL   ((size_t)206000000)    // f32 : knew
#define OFF_NMAP   ((size_t)207000000)    // int : n
#define OFF_CNTA   ((size_t)208000000)    // u32 : 32 (25 attr counts + [25]=total)
#define OFF_MEANE  ((size_t)208000256)    // f32 : 32
#define OFF_EETAB  ((size_t)208000512)    // f32 : 26*128
#define OFF_SCNT   ((size_t)208016000)    // u32 : 5
#define OFF_SSUM   ((size_t)208016064)    // f32 : 5
#define OFF_SFSUM  ((size_t)208016128)    // f32 : 5*64
#define OFF_SFMAX  ((size_t)208017536)    // u32 : 5*64 (ordered-encoded)

// ---------------- helpers ----------------
__device__ __forceinline__ unsigned short f2bf(float f) {
    unsigned u = __float_as_uint(f);
    unsigned r = u + 0x7FFFu + ((u >> 16) & 1u);
    return (unsigned short)(r >> 16);
}
__device__ __forceinline__ unsigned enc_f32(float f) {
    unsigned u = __float_as_uint(f);
    return (u & 0x80000000u) ? ~u : (u | 0x80000000u);
}
__device__ __forceinline__ float dec_f32(unsigned u) {
    unsigned v = (u & 0x80000000u) ? (u & 0x7FFFFFFFu) : ~u;
    return __uint_as_float(v);
}
__device__ __forceinline__ float lrelu(float v) { return v > 0.f ? v : 0.2f * v; }

// ---------------- generic fill ----------------
__global__ void k_fill_u32(unsigned* __restrict__ p, unsigned v, size_t n) {
    size_t i = (size_t)blockIdx.x * blockDim.x + threadIdx.x;
    if (i < n) p[i] = v;
}

// ---------------- edge-state init ----------------
__global__ void k_init_edges(const int* __restrict__ ei, int* __restrict__ src,
                             int* __restrict__ dst, int* __restrict__ alv) {
    int e = blockIdx.x * blockDim.x + threadIdx.x;
    if (e >= EE) return;
    src[e] = ei[e];
    dst[e] = ei[EE + e];
    alv[e] = 1;
}

// ---------------- H build: [x | rwpe | node_emb[attr]] padded ----------------
__global__ void k_build_h(unsigned short* __restrict__ Hb, const float* __restrict__ x, int dimx,
                          const float* __restrict__ rwpe, const int* __restrict__ nattr,
                          const float* __restrict__ nemb, int n, int Kpad) {
    size_t t = (size_t)blockIdx.x * blockDim.x + threadIdx.x;
    if (t >= (size_t)n * Kpad) return;
    int i = (int)(t / Kpad), k = (int)(t % Kpad);
    float v;
    if (k < dimx)            v = x[(size_t)i * dimx + k];
    else if (k < dimx + 16)  v = rwpe[(size_t)i * 16 + (k - dimx)];
    else if (k < dimx + 48)  v = nemb[nattr[i] * 32 + (k - dimx - 16)];
    else                     v = 0.f;
    Hb[t] = f2bf(v);
}

// ---------------- W^T build: WT[j][k] = (j<128 ? Wl : Wr)[k][j] ----------------
__global__ void k_build_wt(unsigned short* __restrict__ WT, const float* __restrict__ Wl,
                           const float* __restrict__ Wr, int din, int Kpad) {
    int t = blockIdx.x * blockDim.x + threadIdx.x;
    if (t >= 256 * Kpad) return;
    int j = t / Kpad, k = t % Kpad;
    float v = 0.f;
    if (k < din) v = (j < 128) ? Wl[(size_t)k * 128 + j] : Wr[(size_t)k * 128 + (j - 128)];
    WT[(size_t)j * Kpad + k] = f2bf(v);
}

// ---------------- WMMA GEMM: C[M,256] = H[M,Kpad] * W[Kpad,256] ----------------
// A tile (16 x KC) staged in LDS once per block, shared by all 8 waves / 16 col tiles.
union FragBF { bf16x16 v; uint4 q[2]; };

__global__ void __launch_bounds__(256) k_gemm_wmma(const unsigned short* __restrict__ Hb,
                                                   const unsigned short* __restrict__ WT,
                                                   float* __restrict__ Cout,
                                                   int M, int Kpad, int Nout) {
    __shared__ unsigned short lds[16 * KC];   // 4 KB A-chunk
    const int tid  = threadIdx.x;
    const int lane = tid & 31;
    const int wave = tid >> 5;
    const int r0 = blockIdx.x * 16;
    const int khalf = (lane >> 4) << 3;       // 0 or 8
    const int m16 = lane & 15;
    // wave handles column tiles {wave, wave+8} of Nout=256
    const int n0 = (wave << 4) + m16;
    const int n1 = ((wave + 8) << 4) + m16;
    const unsigned short* wrow0 = WT + (size_t)n0 * Kpad;
    const unsigned short* wrow1 = WT + (size_t)n1 * Kpad;
    f32x8 acc0 = {0.f, 0.f, 0.f, 0.f, 0.f, 0.f, 0.f, 0.f};
    f32x8 acc1 = {0.f, 0.f, 0.f, 0.f, 0.f, 0.f, 0.f, 0.f};
    // cooperative A loader: each thread moves one uint4 (8 bf16) per chunk
    const int lr  = tid >> 4;                 // row 0..15
    const int lcs = (tid & 15) << 3;          // col 0,8,...,120
    const int lrow = (r0 + lr < M) ? (r0 + lr) : (M - 1);
    const unsigned short* gsrc = Hb + (size_t)lrow * Kpad + lcs;

    for (int kb0 = 0; kb0 < Kpad; kb0 += KC) {
        __syncthreads();
        *(uint4*)(&lds[lr * KC + lcs]) = *(const uint4*)(gsrc + kb0);
        if (kb0 + KC < Kpad)
            __builtin_prefetch((const void*)(gsrc + kb0 + KC), 0, 1);
        __syncthreads();
        const unsigned short* arow = &lds[m16 * KC];
#pragma unroll
        for (int kb = 0; kb < KC; kb += 32) {
            FragBF a, b0, b1;
            a.q[0]  = *(const uint4*)(arow + kb + khalf);        // K+0..7 / K+8..15
            a.q[1]  = *(const uint4*)(arow + kb + khalf + 16);   // K+16..23 / K+24..31
            b0.q[0] = *(const uint4*)(wrow0 + kb0 + kb + khalf);
            b0.q[1] = *(const uint4*)(wrow0 + kb0 + kb + khalf + 16);
            b1.q[0] = *(const uint4*)(wrow1 + kb0 + kb + khalf);
            b1.q[1] = *(const uint4*)(wrow1 + kb0 + kb + khalf + 16);
            acc0 = __builtin_amdgcn_wmma_f32_16x16x32_bf16(
                false, a.v, false, b0.v, (short)0, acc0, false, false);
            acc1 = __builtin_amdgcn_wmma_f32_16x16x32_bf16(
                false, a.v, false, b1.v, (short)0, acc1, false, false);
        }
    }
    const int rbase = r0 + khalf;
#pragma unroll
    for (int i = 0; i < 8; ++i) {
        int r = rbase + i;
        if (r < M) {
            Cout[(size_t)r * Nout + (wave << 4) + m16]       = acc0[i];
            Cout[(size_t)r * Nout + ((wave + 8) << 4) + m16] = acc1[i];
        }
    }
}

// ---------------- alive-edge attr histogram + mean edge embedding ----------------
__global__ void k_attr_hist(unsigned* __restrict__ cnt, const int* __restrict__ eattr,
                            const int* __restrict__ alv) {
    int e = blockIdx.x * blockDim.x + threadIdx.x;
    if (e >= EE || !alv[e]) return;
    atomicAdd(&cnt[eattr[e]], 1u);
    atomicAdd(&cnt[25], 1u);
}
__global__ void k_mean_e(float* __restrict__ me, const unsigned* __restrict__ cnt,
                         const float* __restrict__ eemb) {
    int d = threadIdx.x;
    if (d >= 32) return;
    float tot = fmaxf((float)cnt[25], 1.f);
    float s = 0.f;
    for (int a = 0; a < 25; ++a) s += (float)cnt[a] * eemb[a * 32 + d];
    me[d] = s / tot;
}
__global__ void k_eetab(float* __restrict__ tab, const float* __restrict__ eemb,
                        const float* __restrict__ me, const float* __restrict__ We) {
    int t = blockIdx.x * blockDim.x + threadIdx.x;
    if (t >= 26 * 128) return;
    int a = t / 128, j = t % 128;
    const float* row = (a < 25) ? (eemb + a * 32) : me;
    float s = 0.f;
    for (int d = 0; d < 32; ++d) s += row[d] * We[d * 128 + j];
    tab[t] = s;
}

// ---------------- GATv2 edge passes ----------------
__global__ void k_edge_max(const float* __restrict__ XLR, const float* __restrict__ tab,
                           const int* __restrict__ src, const int* __restrict__ dst,
                           const int* __restrict__ alv, const int* __restrict__ eattr,
                           const float* __restrict__ att, unsigned* __restrict__ mx, int n) {
    int t = blockIdx.x * blockDim.x + threadIdx.x;
    if (t >= EE + n) return;
    int s, d; const float* ee;
    if (t < EE) {
        if (!alv[t]) return;
        s = src[t]; d = dst[t]; ee = tab + eattr[t] * 128;
    } else { s = d = t - EE; ee = tab + 25 * 128; }
    const float* xl = XLR + (size_t)s * 256;
    const float* xr = XLR + (size_t)d * 256 + 128;
#pragma unroll
    for (int h = 0; h < 2; ++h) {
        float lg = 0.f;
        for (int c = 0; c < 64; ++c)
            lg += lrelu(xl[h * 64 + c] + xr[h * 64 + c] + ee[h * 64 + c]) * att[h * 64 + c];
        atomicMax(&mx[(size_t)d * 2 + h], enc_f32(lg));
    }
}

__global__ void k_edge_acc(const float* __restrict__ XLR, const float* __restrict__ tab,
                           const int* __restrict__ src, const int* __restrict__ dst,
                           const int* __restrict__ alv, const int* __restrict__ eattr,
                           const float* __restrict__ att, const unsigned* __restrict__ mx,
                           float* __restrict__ den, float* __restrict__ acc, int n) {
    int t = blockIdx.x * blockDim.x + threadIdx.x;
    if (t >= EE + n) return;
    int s, d; const float* ee;
    if (t < EE) {
        if (!alv[t]) return;
        s = src[t]; d = dst[t]; ee = tab + eattr[t] * 128;
    } else { s = d = t - EE; ee = tab + 25 * 128; }
    const float* xl = XLR + (size_t)s * 256;
    const float* xr = XLR + (size_t)d * 256 + 128;
#pragma unroll
    for (int h = 0; h < 2; ++h) {
        float lg = 0.f;
        for (int c = 0; c < 64; ++c)
            lg += lrelu(xl[h * 64 + c] + xr[h * 64 + c] + ee[h * 64 + c]) * att[h * 64 + c];
        float ex = __expf(lg - dec_f32(mx[(size_t)d * 2 + h]));
        atomicAdd(&den[(size_t)d * 2 + h], ex);
        for (int c = 0; c < 64; ++c)
            atomicAdd(&acc[(size_t)d * 128 + h * 64 + c], ex * xl[h * 64 + c]);
    }
}

__global__ void k_node_final(const float* __restrict__ acc, const float* __restrict__ den,
                             const float* __restrict__ bvec, float* __restrict__ hf, int n) {
    int t = blockIdx.x * blockDim.x + threadIdx.x;
    if (t >= n * 64) return;
    int i = t / 64, c = t % 64;
    float o0 = acc[(size_t)i * 128 + c]      / fmaxf(den[(size_t)i * 2],     1e-16f);
    float o1 = acc[(size_t)i * 128 + 64 + c] / fmaxf(den[(size_t)i * 2 + 1], 1e-16f);
    hf[t] = fmaxf(0.f, 0.5f * (o0 + o1) + bvec[c]);
}

// ---------------- SAGPool: GraphConv score ----------------
__global__ void k_sag_agg(const float* __restrict__ hf, const int* __restrict__ src,
                          const int* __restrict__ dst, const int* __restrict__ alv,
                          float* __restrict__ agg) {
    int e = blockIdx.x * blockDim.x + threadIdx.x;
    if (e >= EE || !alv[e]) return;
    int s = src[e], d = dst[e];
    for (int c = 0; c < 64; ++c)
        atomicAdd(&agg[(size_t)d * 64 + c], hf[(size_t)s * 64 + c]);
}
__global__ void k_score(const float* __restrict__ agg, const float* __restrict__ hf,
                        const float* __restrict__ prel, const float* __restrict__ pbias,
                        const float* __restrict__ proot, float* __restrict__ score, int n) {
    int i = blockIdx.x * blockDim.x + threadIdx.x;
    if (i >= n) return;
    float s = pbias[0];
    for (int c = 0; c < 64; ++c)
        s += agg[(size_t)i * 64 + c] * prel[c] + hf[(size_t)i * 64 + c] * proot[c];
    score[i] = tanhf(s);
}

// ---------------- top-k via bitonic sort of (~enc(score)<<32 | idx) ----------------
__global__ void k_keys(unsigned long long* __restrict__ keys, const float* __restrict__ score, int n) {
    int i = blockIdx.x * blockDim.x + threadIdx.x;
    if (i >= SORTP) return;
    keys[i] = (i < n) ? (((unsigned long long)(~enc_f32(score[i])) << 32) | (unsigned)i)
                      : ~0ull;
}
__global__ void k_bitonic(unsigned long long* __restrict__ keys, unsigned j, unsigned k) {
    unsigned i = blockIdx.x * blockDim.x + threadIdx.x;
    unsigned ixj = i ^ j;
    if (ixj <= i) return;
    bool up = ((i & k) == 0);
    unsigned long long a = keys[i], b = keys[ixj];
    if ((a > b) == up) { keys[i] = b; keys[ixj] = a; }
}
__global__ void k_extract(const unsigned long long* __restrict__ keys,
                          const float* __restrict__ score,
                          int* __restrict__ perm, float* __restrict__ sval, int knew) {
    int j = blockIdx.x * blockDim.x + threadIdx.x;
    if (j >= knew) return;
    int idx = (int)(keys[j] & 0xffffffffull);
    perm[j] = idx;
    sval[j] = score[idx];
}
__global__ void k_nmap_init(int* __restrict__ nmap, int n) {
    int i = blockIdx.x * blockDim.x + threadIdx.x;
    if (i < n) nmap[i] = -1;
}
__global__ void k_nmap_set(int* __restrict__ nmap, const int* __restrict__ perm, int knew) {
    int j = blockIdx.x * blockDim.x + threadIdx.x;
    if (j < knew) nmap[perm[j]] = j;
}
__global__ void k_remap(int* __restrict__ src, int* __restrict__ dst, int* __restrict__ alv,
                        const int* __restrict__ nmap) {
    int e = blockIdx.x * blockDim.x + threadIdx.x;
    if (e >= EE) return;
    int a = alv[e];
    int ns = nmap[src[e]], nd = nmap[dst[e]];
    int ok = (a && ns >= 0 && nd >= 0) ? 1 : 0;
    src[e] = ok ? ns : 0;
    dst[e] = ok ? nd : 0;
    alv[e] = ok;
}
__global__ void k_pool_x(const float* __restrict__ hf, const int* __restrict__ perm,
                         const float* __restrict__ sval, float* __restrict__ xn, int knew) {
    int t = blockIdx.x * blockDim.x + threadIdx.x;
    if (t >= knew * 64) return;
    int j = t / 64, c = t % 64;
    xn[t] = hf[(size_t)perm[j] * 64 + c] * sval[j];
}
__global__ void k_pool_misc(const int* __restrict__ perm, const float* __restrict__ rwpe0,
                            const int* __restrict__ nattr0, float* __restrict__ rwpe1,
                            int* __restrict__ nattr1, int knew) {
    int j = blockIdx.x * blockDim.x + threadIdx.x;
    if (j >= knew) return;
    int p = perm[j];
    nattr1[j] = nattr0[p];
    for (int w = 0; w < 16; ++w) rwpe1[(size_t)j * 16 + w] = rwpe0[(size_t)p * 16 + w];
}

// ---------------- stain pooling ----------------
__global__ void k_stain_acc(const float* __restrict__ xn, const int* __restrict__ perm,
                            const float* __restrict__ sval, const int* __restrict__ nattr_orig,
                            unsigned* __restrict__ scnt, float* __restrict__ ssum,
                            float* __restrict__ sfsum, unsigned* __restrict__ sfmax, int knew) {
    int j = blockIdx.x * blockDim.x + threadIdx.x;
    if (j >= knew) return;
    int s = nattr_orig[perm[j]];   // faithful: ORIGINAL node_attr indexed by local perm
    atomicAdd(&scnt[s], 1u);
    atomicAdd(&ssum[s], sval[j]);
    for (int c = 0; c < 64; ++c) {
        float v = xn[(size_t)j * 64 + c];
        atomicAdd(&sfsum[s * 64 + c], v);
        atomicMax(&sfmax[s * 64 + c], enc_f32(v));
    }
}
__global__ void k_stain_final(const unsigned* __restrict__ scnt, const float* __restrict__ ssum,
                              const float* __restrict__ sfsum, const unsigned* __restrict__ sfmax,
                              float* __restrict__ out) {
    int c = threadIdx.x;
    if (c >= 128) return;
    float ws[5], wsum = 0.f;
    for (int s = 0; s < 5; ++s) {
        float cnt = (float)scnt[s];
        float w = (cnt > 0.f) ? (ssum[s] / fmaxf(cnt, 1.f)) : 0.f;
        ws[s] = w; wsum += w;
    }
    float r = 0.f;
    if (c < 64) {
        for (int s = 0; s < 5; ++s)
            r += (ws[s] / wsum) * (sfsum[s * 64 + c] / fmaxf((float)scnt[s], 1.f));
    } else {
        int cc = c - 64;
        for (int s = 0; s < 5; ++s) {
            float fm = (scnt[s] > 0u) ? dec_f32(sfmax[s * 64 + cc]) : 0.f;
            r += (ws[s] / wsum) * fm;
        }
    }
    out[c] = r;
}

// ---------------- host-side layer driver ----------------
static inline int gridFor(size_t n, int b) { return (int)((n + (size_t)b - 1) / (size_t)b); }

static void run_layer(hipStream_t stream, char* ws, float* out, int layer,
                      int n, int knew, int dimx, int din, int Kpad,
                      const float* xsrc, const float* rwpe_src, const int* nattr_src,
                      const float* nemb, const float* eemb, const int* eattr,
                      const int* nattr_orig,
                      const float* Wl, const float* Wr, const float* We, const float* att,
                      const float* bvec, const float* prel, const float* pbias,
                      const float* proot,
                      float* xnew_out, float* rwpe_new, int* nattr_new) {
    unsigned short* HB  = (unsigned short*)(ws + OFF_HB);
    unsigned short* WT  = (unsigned short*)(ws + OFF_WT);
    float*    XLR  = (float*)(ws + OFF_XLR);
    float*    ACC  = (float*)(ws + OFF_ACC);
    float*    HF   = (float*)(ws + OFF_HF);
    float*    AGG  = (float*)(ws + OFF_AGG);
    int*      SRC  = (int*)(ws + OFF_SRC);
    int*      DST  = (int*)(ws + OFF_DST);
    int*      ALV  = (int*)(ws + OFF_ALV);
    unsigned* MX   = (unsigned*)(ws + OFF_MX);
    float*    DEN  = (float*)(ws + OFF_DEN);
    float*    SCORE= (float*)(ws + OFF_SCORE);
    unsigned long long* KEYS = (unsigned long long*)(ws + OFF_KEYS);
    int*      PERM = (int*)(ws + OFF_PERM);
    float*    SVAL = (float*)(ws + OFF_SVAL);
    int*      NMAP = (int*)(ws + OFF_NMAP);
    unsigned* CNTA = (unsigned*)(ws + OFF_CNTA);
    float*    MEANE= (float*)(ws + OFF_MEANE);
    float*    EETAB= (float*)(ws + OFF_EETAB);
    unsigned* SCNT = (unsigned*)(ws + OFF_SCNT);
    float*    SSUM = (float*)(ws + OFF_SSUM);
    float*    SFSUM= (float*)(ws + OFF_SFSUM);
    unsigned* SFMAX= (unsigned*)(ws + OFF_SFMAX);

    // 1) projections via WMMA
    k_build_wt<<<gridFor(256 * Kpad, 256), 256, 0, stream>>>(WT, Wl, Wr, din, Kpad);
    k_build_h<<<gridFor((size_t)n * Kpad, 256), 256, 0, stream>>>(HB, xsrc, dimx, rwpe_src,
                                                                  nattr_src, nemb, n, Kpad);
    k_gemm_wmma<<<gridFor(n, 16), 256, 0, stream>>>(HB, WT, XLR, n, Kpad, 256);

    // 2) edge embedding table (25 attrs + mean row)
    k_fill_u32<<<1, 32, 0, stream>>>(CNTA, 0u, 32);
    k_attr_hist<<<gridFor(EE, 256), 256, 0, stream>>>(CNTA, eattr, ALV);
    k_mean_e<<<1, 32, 0, stream>>>(MEANE, CNTA, eemb);
    k_eetab<<<gridFor(26 * 128, 256), 256, 0, stream>>>(EETAB, eemb, MEANE, We);

    // 3) GATv2 segment softmax + aggregation
    k_fill_u32<<<gridFor((size_t)n * 2, 256), 256, 0, stream>>>(MX, 0u, (size_t)n * 2);
    k_fill_u32<<<gridFor((size_t)n * 2, 256), 256, 0, stream>>>((unsigned*)DEN, 0u, (size_t)n * 2);
    k_fill_u32<<<gridFor((size_t)n * 128, 256), 256, 0, stream>>>((unsigned*)ACC, 0u, (size_t)n * 128);
    k_edge_max<<<gridFor(EE + n, 256), 256, 0, stream>>>(XLR, EETAB, SRC, DST, ALV, eattr, att, MX, n);
    k_edge_acc<<<gridFor(EE + n, 256), 256, 0, stream>>>(XLR, EETAB, SRC, DST, ALV, eattr, att,
                                                          MX, DEN, ACC, n);
    k_node_final<<<gridFor((size_t)n * 64, 256), 256, 0, stream>>>(ACC, DEN, bvec, HF, n);

    // 4) SAGPool score
    k_fill_u32<<<gridFor((size_t)n * 64, 256), 256, 0, stream>>>((unsigned*)AGG, 0u, (size_t)n * 64);
    k_sag_agg<<<gridFor(EE, 256), 256, 0, stream>>>(HF, SRC, DST, ALV, AGG);
    k_score<<<gridFor(n, 256), 256, 0, stream>>>(AGG, HF, prel, pbias, proot, SCORE, n);

    // 5) exact top-k (bitonic sort)
    k_keys<<<gridFor(SORTP, 256), 256, 0, stream>>>(KEYS, SCORE, n);
    for (unsigned k = 2; k <= SORTP; k <<= 1)
        for (unsigned j = k >> 1; j > 0; j >>= 1)
            k_bitonic<<<SORTP / 256, 256, 0, stream>>>(KEYS, j, k);
    k_extract<<<gridFor(knew, 256), 256, 0, stream>>>(KEYS, SCORE, PERM, SVAL, knew);

    // 6) remap + pooled features
    k_nmap_init<<<gridFor(n, 256), 256, 0, stream>>>(NMAP, n);
    k_nmap_set<<<gridFor(knew, 256), 256, 0, stream>>>(NMAP, PERM, knew);
    k_remap<<<gridFor(EE, 256), 256, 0, stream>>>(SRC, DST, ALV, NMAP);
    k_pool_x<<<gridFor((size_t)knew * 64, 256), 256, 0, stream>>>(HF, PERM, SVAL, xnew_out, knew);
    if (rwpe_new)
        k_pool_misc<<<gridFor(knew, 256), 256, 0, stream>>>(PERM, rwpe_src, nattr_src,
                                                            rwpe_new, nattr_new, knew);

    // 7) stain pooling -> out[layer*128 .. +128)
    k_fill_u32<<<1, 32, 0, stream>>>(SCNT, 0u, 5);
    k_fill_u32<<<1, 32, 0, stream>>>((unsigned*)SSUM, 0u, 5);
    k_fill_u32<<<gridFor(320, 256), 256, 0, stream>>>((unsigned*)SFSUM, 0u, 320);
    k_fill_u32<<<gridFor(320, 256), 256, 0, stream>>>(SFMAX, 0u, 320);
    k_stain_acc<<<gridFor(knew, 256), 256, 0, stream>>>(xnew_out, PERM, SVAL, nattr_orig,
                                                        SCNT, SSUM, SFSUM, SFMAX, knew);
    k_stain_final<<<1, 128, 0, stream>>>(SCNT, SSUM, SFSUM, SFMAX, out + layer * 128);
}

extern "C" void kernel_launch(void* const* d_in, const int* in_sizes, int n_in,
                              void* d_out, int out_size, void* d_ws, size_t ws_size,
                              hipStream_t stream) {
    const float* x      = (const float*)d_in[0];
    const int*   ei     = (const int*)d_in[1];
    const int*   eattr  = (const int*)d_in[2];
    const int*   nattr  = (const int*)d_in[3];
    const float* rwpe   = (const float*)d_in[4];
    const float* nemb   = (const float*)d_in[6];
    const float* eemb   = (const float*)d_in[7];
    const float* Wl0 = (const float*)d_in[8],  *Wr0 = (const float*)d_in[9];
    const float* We0 = (const float*)d_in[10], *att0 = (const float*)d_in[11];
    const float* b0  = (const float*)d_in[12], *prel0 = (const float*)d_in[13];
    const float* pb0 = (const float*)d_in[14], *proot0 = (const float*)d_in[15];
    const float* Wl1 = (const float*)d_in[16], *Wr1 = (const float*)d_in[17];
    const float* We1 = (const float*)d_in[18], *att1 = (const float*)d_in[19];
    const float* b1  = (const float*)d_in[20], *prel1 = (const float*)d_in[21];
    const float* pb1 = (const float*)d_in[22], *proot1 = (const float*)d_in[23];

    char* ws = (char*)d_ws;
    float* out = (float*)d_out;

    int*   SRC  = (int*)(ws + OFF_SRC);
    int*   DST  = (int*)(ws + OFF_DST);
    int*   ALV  = (int*)(ws + OFF_ALV);
    float* XN1  = (float*)(ws + OFF_XN1);
    float* XN2  = (float*)(ws + OFF_XN2);
    float* RW1  = (float*)(ws + OFF_RWPE1);
    int*   NAT1 = (int*)(ws + OFF_NAT1);

    k_init_edges<<<gridFor(EE, 256), 256, 0, stream>>>(ei, SRC, DST, ALV);

    // layer 0: n=50000 -> k=35000
    run_layer(stream, ws, out, 0,
              NN, K0N, 1024, DIN0, PAD0,
              x, rwpe, nattr,
              nemb, eemb, eattr, nattr,
              Wl0, Wr0, We0, att0, b0, prel0, pb0, proot0,
              XN1, RW1, NAT1);

    // layer 1: n=35000 -> k=24500
    run_layer(stream, ws, out, 1,
              K0N, K1N, 64, DIN1, PAD1,
              XN1, RW1, NAT1,
              nemb, eemb, eattr, nattr,
              Wl1, Wr1, We1, att1, b1, prel1, pb1, proot1,
              XN2, nullptr, nullptr);
}
